// CollectiveAgent_86887188398364
// MI455X (gfx1250) — matrix operand order repaired and checked by reading, not compiled, not run on hardware
//
#include <hip/hip_runtime.h>
#include <hip/hip_bf16.h>
#include <cstddef>
#include <cstdint>

#define AS3 __attribute__((address_space(3)))

// ---------------------------------------------------------------------------
// Types for CDNA5 WMMA
// ---------------------------------------------------------------------------
typedef __attribute__((ext_vector_type(16))) __bf16          v16bf;
typedef __attribute__((ext_vector_type(8)))  float           v8f;
typedef __attribute__((ext_vector_type(8)))  unsigned short  u16x8;
typedef __attribute__((ext_vector_type(16))) unsigned short  u16x16;

#define BATCH 4096

__device__ __forceinline__ unsigned short f2bf_u(float f) {
  unsigned u = __builtin_bit_cast(unsigned, f);
  unsigned r = u + 0x7FFFu + ((u >> 16) & 1u);   // round-to-nearest-even
  return (unsigned short)(r >> 16);
}
__device__ __forceinline__ float bf2f(unsigned short h) {
  unsigned u = (unsigned)h << 16;
  return __builtin_bit_cast(float, u);
}

__device__ __forceinline__ float wsum(float v) {
#pragma unroll
  for (int o = 16; o > 0; o >>= 1) v += __shfl_xor(v, o, 32);
  return v;
}

// Async global->LDS copy of 16 bytes per lane (gfx1250 VGLOBAL async encoding,
// tracked by ASYNCcnt). vdst = per-lane LDS byte address, vaddr = 64-bit addr.
__device__ __forceinline__ void async_copy_b128(const void* gptr, void* lptr) {
  unsigned lds_addr = (unsigned)(size_t)(AS3 char*)(char*)lptr;
  unsigned long long gaddr = (unsigned long long)(size_t)gptr;
  asm volatile("global_load_async_to_lds_b128 %0, %1, off"
               :: "v"(lds_addr), "v"(gaddr) : "memory");
}
__device__ __forceinline__ void async_wait0() {
  asm volatile("s_wait_asynccnt 0x0" ::: "memory");
}

// ---------------------------------------------------------------------------
// bf16 WMMA GEMM: C[M,N] = act(A[M,K] @ Bt^T + bias)
//   A  : M x K bf16 row-major        (lda = row stride, elements)
//   Bt : N x K bf16 row-major        (ldb = row stride) -- weights pre-transposed
//   ACT: 0 = none, 1 = tanh ; OUTBF: 0 = f32 C, 1 = bf16 C
// Requires M%256==0, N%64==0, K%32==0 (true for every call here).
// Block: 256 threads (8 waves). Tile: 256(M) x 64(N), K step 32.
// Tiles staged entirely with global_load_async_to_lds_b128 (no VALU, no VGPR
// round-trip); each wave runs 8 back-to-back v_wmma_f32_16x16x32_bf16.
// ---------------------------------------------------------------------------
template <int ACT, int OUTBF>
__global__ __launch_bounds__(256)
void gemm_wmma_bf16(const unsigned short* __restrict__ A, int lda,
                    const unsigned short* __restrict__ Bt, int ldb,
                    const float* __restrict__ bias,
                    void* __restrict__ Cv, int ldc,
                    int M, int N, int K) {
  __shared__ __attribute__((aligned(16))) unsigned short As[256][40];
  __shared__ __attribute__((aligned(16))) unsigned short Bs[64][40];

  const int tid  = threadIdx.x;
  const int wave = tid >> 5;
  const int lane = tid & 31;
  const int g    = lane >> 4;     // half-wave group
  const int l4   = lane & 15;
  const int m0   = blockIdx.y * 256;
  const int n0   = blockIdx.x * 64;
  const int brow = tid >> 2;            // B tile: 64 rows x 4 x 16B
  const int bc8  = (tid & 3) * 8;

  v8f acc[2][4];
#pragma unroll
  for (int p = 0; p < 2; ++p)
#pragma unroll
    for (int j = 0; j < 4; ++j) acc[p][j] = v8f{};

  for (int k0 = 0; k0 < K; k0 += 32) {
    // ---- async stage A tile 256x32 bf16: 1024 x 16B, 4 per thread ----
#pragma unroll
    for (int i = 0; i < 4; ++i) {
      int idx = tid + i * 256;
      int row = idx >> 2;
      int jj  = (idx & 3) * 8;
      async_copy_b128(A + (size_t)(m0 + row) * lda + k0 + jj, &As[row][jj]);
    }
    // ---- async stage B tile 64x32 bf16: 256 x 16B, 1 per thread ----
    async_copy_b128(Bt + (size_t)(n0 + brow) * ldb + k0 + bc8, &Bs[brow][bc8]);
    // ---- prefetch next K panel into L2 ----
    if (k0 + 32 < K) {
      __builtin_prefetch(A + (size_t)(m0 + tid) * lda + k0 + 32, 0, 3);
      __builtin_prefetch(Bt + (size_t)(n0 + (tid & 63)) * ldb + k0 + 32, 0, 3);
    }
    async_wait0();
    __syncthreads();

    // ---- load ALL fragments (ISA 16-bit A 16x32 / B 32x16 layouts) ----
    v16bf af[2];
#pragma unroll
    for (int p = 0; p < 2; ++p) {
      const int arow = wave * 32 + p * 16 + l4;
      u16x8 lo = *(const u16x8*)&As[arow][8 * g];
      u16x8 hi = *(const u16x8*)&As[arow][16 + 8 * g];
      u16x16 fu;
#pragma unroll
      for (int i = 0; i < 8; ++i) { fu[i] = lo[i]; fu[8 + i] = hi[i]; }
      af[p] = __builtin_bit_cast(v16bf, fu);
    }
    v16bf bf[4];
#pragma unroll
    for (int j = 0; j < 4; ++j) {
      const int bcol = j * 16 + l4;
      u16x8 lo = *(const u16x8*)&Bs[bcol][8 * g];
      u16x8 hi = *(const u16x8*)&Bs[bcol][16 + 8 * g];
      u16x16 fu;
#pragma unroll
      for (int i = 0; i < 8; ++i) { fu[i] = lo[i]; fu[8 + i] = hi[i]; }
      bf[j] = __builtin_bit_cast(v16bf, fu);
    }
    // ---- 8 back-to-back WMMAs ----
#pragma unroll
    for (int j = 0; j < 4; ++j)
#pragma unroll
      for (int p = 0; p < 2; ++p)
        acc[p][j] = __builtin_amdgcn_wmma_f32_16x16x32_bf16(
            false, af[p], false, bf[j], (short)0, acc[p][j], false, false);
    __syncthreads();
  }

  // ---- epilogue ----
#pragma unroll
  for (int j = 0; j < 4; ++j) {
    int n = n0 + j * 16 + l4;
    float bv = bias ? bias[n] : 0.0f;
#pragma unroll
    for (int p = 0; p < 2; ++p) {
#pragma unroll
      for (int r = 0; r < 8; ++r) {
        int m = m0 + wave * 32 + p * 16 + (g << 3) + r;
        float v = acc[p][j][r] + bv;
        if (ACT == 1) v = tanhf(v);
        if (OUTBF) ((unsigned short*)Cv)[(size_t)m * ldc + n] = f2bf_u(v);
        else       ((float*)Cv)[(size_t)m * ldc + n] = v;
      }
    }
  }
}

// ---------------------------------------------------------------------------
// Reassociated attention: per batch b, per head h (1 block of 4 waves per b)
//   score[s] = (u_bh . mem_bs + q_bh . bk_h) / 16 ;  softmax over 16 slots
//   MIX[h,b,:] = sum_s attn_s * mem_bs (bf16 out; MIX may alias U)
// 64KB f32 memory tile staged with async global->LDS.
// ---------------------------------------------------------------------------
__global__ __launch_bounds__(128)
void attn_mix_kernel(const float* __restrict__ mem,            // (B,16,1024) f32
                     const unsigned short* __restrict__ Q,     // (B,1024) bf16
                     const unsigned short* __restrict__ U,     // (4,B,1024) bf16
                     const float* __restrict__ bk,             // (1024) f32
                     unsigned short* __restrict__ MIX) {       // (4,B,1024) bf16
  const int b    = blockIdx.x;
  const int tid  = threadIdx.x;
  const int h    = tid >> 5;
  const int lane = tid & 31;
  __shared__ __attribute__((aligned(16))) float sm[16 * 1024];   // 64 KB

  const float* mb = mem + (size_t)b * 16 * 1024;
#pragma unroll
  for (int i = 0; i < 32; ++i) {
    int idx = tid + i * 128;
    async_copy_b128(mb + idx * 4, sm + idx * 4);
  }
  async_wait0();
  __syncthreads();

  const unsigned short* q = Q + (size_t)b * 1024 + h * 256;
  const unsigned short* u = U + ((size_t)h * gridDim.x + b) * 1024;

  float qb = 0.0f;
  for (int e = lane; e < 256; e += 32) qb += bf2f(q[e]) * bk[h * 256 + e];
  qb = wsum(qb);

  float sc[16];
#pragma unroll
  for (int s = 0; s < 16; ++s) sc[s] = 0.0f;
  for (int e = lane; e < 1024; e += 32) {
    float uv = bf2f(u[e]);
#pragma unroll
    for (int s = 0; s < 16; ++s) sc[s] += uv * sm[s * 1024 + e];
  }
#pragma unroll
  for (int s = 0; s < 16; ++s) sc[s] = wsum(sc[s]);

  float mx = -1e30f;
#pragma unroll
  for (int s = 0; s < 16; ++s) { sc[s] = (sc[s] + qb) * 0.0625f; mx = fmaxf(mx, sc[s]); }
  float den = 0.0f;
#pragma unroll
  for (int s = 0; s < 16; ++s) { sc[s] = expf(sc[s] - mx); den += sc[s]; }
  float inv = 1.0f / den;

  unsigned short* mo = MIX + ((size_t)h * gridDim.x + b) * 1024;
  for (int e = lane; e < 1024; e += 32) {
    float a = 0.0f;
#pragma unroll
    for (int s = 0; s < 16; ++s) a += sc[s] * sm[s * 1024 + e];
    mo[e] = f2bf_u(a * inv);
  }
}

// ---------------------------------------------------------------------------
// Pre-pass conversion kernels
// ---------------------------------------------------------------------------
// f32 (K x N) -> bf16 transposed (N x K)
__global__ void tconv_kernel(const float* __restrict__ in,
                             unsigned short* __restrict__ out,
                             int K, int N, int total) {
  int i = blockIdx.x * blockDim.x + threadIdx.x;
  if (i >= total) return;
  int n = i / K, k = i - n * K;
  out[(size_t)n * K + k] = f2bf_u(in[(size_t)k * N + n]);
}
// f32 -> bf16, same layout
__global__ void cvt_kernel(const float* __restrict__ in,
                           unsigned short* __restrict__ out, int n) {
  int i = blockIdx.x * blockDim.x + threadIdx.x;
  if (i < n) out[i] = f2bf_u(in[i]);
}

// ---------------------------------------------------------------------------
// Fused elementwise kernels (f32 math, bf16 outputs where a GEMM consumes)
// ---------------------------------------------------------------------------
__global__ void sub_bf_kernel(const float* a, const float* b, unsigned short* o, int n) {
  int i = blockIdx.x * blockDim.x + threadIdx.x;
  if (i < n) o[i] = f2bf_u(a[i] - b[i]);
}
__global__ void axpy_bf_kernel(const float* a, const unsigned short* b, float s,
                               unsigned short* o, int n) {
  int i = blockIdx.x * blockDim.x + threadIdx.x;
  if (i < n) o[i] = f2bf_u(a[i] + s * bf2f(b[i]));
}
__global__ void concat_fb_kernel(const float* a, int wa, const unsigned short* b, int wb,
                                 unsigned short* o, int n) {
  int i = blockIdx.x * blockDim.x + threadIdx.x;
  if (i >= n) return;
  int w = wa + wb, r = i / w, c = i - r * w;
  o[i] = (c < wa) ? f2bf_u(a[(size_t)r * wa + c]) : b[(size_t)r * wb + (c - wa)];
}
__global__ void concat_bb_kernel(const unsigned short* a, int wa,
                                 const unsigned short* b, int wb,
                                 unsigned short* o, int n) {
  int i = blockIdx.x * blockDim.x + threadIdx.x;
  if (i >= n) return;
  int w = wa + wb, r = i / w, c = i - r * w;
  o[i] = (c < wa) ? a[(size_t)r * wa + c] : b[(size_t)r * wb + (c - wa)];
}
__global__ void gru_kernel(const float* __restrict__ gi, const float* __restrict__ gh,
                           const float* __restrict__ h, unsigned short* __restrict__ o,
                           int n) {
  int i = blockIdx.x * blockDim.x + threadIdx.x;
  if (i >= n) return;
  int b = i >> 10, c = i & 1023;
  size_t base = (size_t)b * 3072 + c;
  float r = 1.0f / (1.0f + expf(-(gi[base] + gh[base])));
  float z = 1.0f / (1.0f + expf(-(gi[base + 1024] + gh[base + 1024])));
  float nn = tanhf(gi[base + 2048] + r * gh[base + 2048]);
  float hv = h[i];
  float bi = (1.0f - z) * nn + z * hv;
  o[i] = f2bf_u(0.7f * hv + 0.3f * bi);   // STABLE mode, update_strength 0.3
}
__global__ void time_emb_kernel(const float* __restrict__ phase,
                                const float* __restrict__ Wph,
                                const float* __restrict__ bph,
                                const float* __restrict__ Ecyc,
                                const int* __restrict__ cyc,
                                unsigned short* __restrict__ o, int n) {
  int i = blockIdx.x * blockDim.x + threadIdx.x;
  if (i >= n) return;
  int b = i >> 7, c = i & 127;
  o[i] = f2bf_u(phase[b * 2] * Wph[c] + phase[b * 2 + 1] * Wph[128 + c] + bph[c] +
                Ecyc[(size_t)cyc[b] * 128 + c]);
}

// ---------------------------------------------------------------------------
// Host-side dispatch helpers
// ---------------------------------------------------------------------------
static inline void launch_gemm(hipStream_t s, const unsigned short* A, int lda,
                               const unsigned short* Bt, int ldb, const float* bias,
                               void* C, int ldc, int M, int N, int K,
                               int act, int outbf) {
  dim3 grd(N / 64, M / 256), blk(256);
  if (act) {
    if (outbf) gemm_wmma_bf16<1, 1><<<grd, blk, 0, s>>>(A, lda, Bt, ldb, bias, C, ldc, M, N, K);
    else       gemm_wmma_bf16<1, 0><<<grd, blk, 0, s>>>(A, lda, Bt, ldb, bias, C, ldc, M, N, K);
  } else {
    if (outbf) gemm_wmma_bf16<0, 1><<<grd, blk, 0, s>>>(A, lda, Bt, ldb, bias, C, ldc, M, N, K);
    else       gemm_wmma_bf16<0, 0><<<grd, blk, 0, s>>>(A, lda, Bt, ldb, bias, C, ldc, M, N, K);
  }
}

extern "C" void kernel_launch(void* const* d_in, const int* in_sizes, int n_in,
                              void* d_out, int out_size, void* d_ws, size_t ws_size,
                              hipStream_t stream) {
  (void)in_sizes; (void)n_in; (void)out_size; (void)ws_size;
  const int B = BATCH;
  const float* obs   = (const float*)d_in[0];
  const float* wst   = (const float*)d_in[1];
  const float* ist   = (const float*)d_in[2];
  const float* wmem  = (const float*)d_in[3];
  const float* imem  = (const float*)d_in[4];
  const float* phase = (const float*)d_in[5];
  // 6:We 7:be and 20..23 (world GRU) are dead: new_world is never returned.
  const float* Wp1 = (const float*)d_in[8];  const float* bp1 = (const float*)d_in[9];
  const float* Wp2 = (const float*)d_in[10]; const float* bp2 = (const float*)d_in[11];
  const float* Wqw = (const float*)d_in[12]; const float* bqw = (const float*)d_in[13];
  const float* Wkw = (const float*)d_in[14]; const float* bkw = (const float*)d_in[15];
  const float* Wvw = (const float*)d_in[16]; const float* bvw = (const float*)d_in[17];
  const float* Wow = (const float*)d_in[18]; const float* bow = (const float*)d_in[19];
  const float* Wep = (const float*)d_in[24]; const float* bep = (const float*)d_in[25];
  const float* Wqi = (const float*)d_in[26]; const float* bqi = (const float*)d_in[27];
  const float* Wki = (const float*)d_in[28]; const float* bki = (const float*)d_in[29];
  const float* Wvi = (const float*)d_in[30]; const float* bvi = (const float*)d_in[31];
  const float* Woi = (const float*)d_in[32]; const float* boi = (const float*)d_in[33];
  const float* Wii = (const float*)d_in[34]; const float* Whi = (const float*)d_in[35];
  const float* bii = (const float*)d_in[36]; const float* bhi = (const float*)d_in[37];
  const float* Wa1 = (const float*)d_in[38]; const float* ba1 = (const float*)d_in[39];
  const float* Wa2 = (const float*)d_in[40]; const float* ba2 = (const float*)d_in[41];
  const float* Wph = (const float*)d_in[42]; const float* bph = (const float*)d_in[43];
  const float* Ecyc = (const float*)d_in[44];
  const int*   cyc  = (const int*)d_in[45];

  // ---------------- workspace layout (bytes, 256B-aligned chunks) ----------
  char* p = (char*)d_ws;
  auto alloc = [&](size_t bytes) -> void* {
    void* r = (void*)p; p += (bytes + 255) & ~(size_t)255; return r;
  };
  typedef unsigned short u16;
  const size_t E = 1024;
  // bf16 weights (transposed to N x K unless noted)
  u16* Wqw_t = (u16*)alloc(E * E * 2);
  u16* Wkw_c = (u16*)alloc(E * E * 2);         // kept (rows = u-output n)
  u16* Wvw_t = (u16*)alloc(E * E * 2);
  u16* Wow_t = (u16*)alloc(E * E * 2);
  u16* Wqi_t = (u16*)alloc(E * E * 2);
  u16* Wki_c = (u16*)alloc(E * E * 2);
  u16* Wvi_t = (u16*)alloc(E * E * 2);
  u16* Woi_t = (u16*)alloc(E * E * 2);
  u16* Wp1_t = (u16*)alloc(1024 * 2048 * 2);
  u16* Wp2_t = (u16*)alloc(512 * 1024 * 2);
  u16* Wep_t = (u16*)alloc(1024 * 512 * 2);
  u16* Wii_t = (u16*)alloc(3072 * 1024 * 2);
  u16* Whi_t = (u16*)alloc(3072 * 1024 * 2);
  u16* Wa1_t = (u16*)alloc(1024 * 1152 * 2);
  u16* Wa2_t = (u16*)alloc(64 * 1024 * 2);
  // bf16 activations
  u16* wst_bf = (u16*)alloc((size_t)B * 1024 * 2);
  u16* ist_bf = (u16*)alloc((size_t)B * 1024 * 2);
  u16* Q_bf   = (u16*)alloc((size_t)B * 1024 * 2);
  u16* U_bf   = (u16*)alloc((size_t)4 * B * 1024 * 2);  // MIX aliased onto U
  u16* CTX_bf = (u16*)alloc((size_t)B * 1024 * 2);
  u16* ATT_bf = (u16*)alloc((size_t)B * 1024 * 2);
  u16* CMB_bf = (u16*)alloc((size_t)B * 2048 * 2);
  u16* P1H_bf = (u16*)alloc((size_t)B * 1024 * 2);
  u16* ERR_bf = (u16*)alloc((size_t)B * 512 * 2);
  u16* II_bf  = (u16*)alloc((size_t)B * 1024 * 2);
  u16* NWI_bf = (u16*)alloc((size_t)B * 1024 * 2);
  u16* TMB_bf = (u16*)alloc((size_t)B * 128 * 2);
  u16* AHN_bf = (u16*)alloc((size_t)B * 1152 * 2);
  u16* HB_bf  = (u16*)alloc((size_t)B * 1024 * 2);
  // f32 intermediates (consumed by f32 elementwise math)
  float* PRED = (float*)alloc((size_t)B * 512 * 4);
  float* ENC  = (float*)alloc((size_t)B * 1024 * 4);
  float* GI   = (float*)alloc((size_t)B * 3072 * 4);
  float* GH   = (float*)alloc((size_t)B * 3072 * 4);

  const int EW = 256;
  auto nb = [](int n) { return (n + 255) / 256; };
  auto tconv = [&](const float* in, u16* out, int K, int N) {
    int total = K * N;
    tconv_kernel<<<nb(total), EW, 0, stream>>>(in, out, K, N, total);
  };
  auto cvt = [&](const float* in, u16* out, int n) {
    cvt_kernel<<<nb(n), EW, 0, stream>>>(in, out, n);
  };

  // ================= pre-pass: weights/states to bf16 ======================
  tconv(Wqw, Wqw_t, 1024, 1024);  cvt(Wkw, Wkw_c, 1024 * 1024);
  tconv(Wvw, Wvw_t, 1024, 1024);  tconv(Wow, Wow_t, 1024, 1024);
  tconv(Wqi, Wqi_t, 1024, 1024);  cvt(Wki, Wki_c, 1024 * 1024);
  tconv(Wvi, Wvi_t, 1024, 1024);  tconv(Woi, Woi_t, 1024, 1024);
  tconv(Wp1, Wp1_t, 2048, 1024);  tconv(Wp2, Wp2_t, 1024, 512);
  tconv(Wep, Wep_t, 512, 1024);
  tconv(Wii, Wii_t, 1024, 3072);  tconv(Whi, Whi_t, 1024, 3072);
  tconv(Wa1, Wa1_t, 1152, 1024);  tconv(Wa2, Wa2_t, 1024, 64);
  cvt(wst, wst_bf, B * 1024);     cvt(ist, ist_bf, B * 1024);

  // ================= world-memory attention (reassociated) =================
  launch_gemm(stream, wst_bf, 1024, Wqw_t, 1024, bqw, Q_bf, 1024, B, 1024, 1024, 0, 1);
  for (int h = 0; h < 4; ++h)
    launch_gemm(stream, Q_bf + h * 256, 1024, Wkw_c + h * 256, 1024, nullptr,
                U_bf + (size_t)h * B * 1024, 1024, B, 1024, 256, 0, 1);
  attn_mix_kernel<<<B, 128, 0, stream>>>(wmem, Q_bf, U_bf, bkw, U_bf);
  for (int h = 0; h < 4; ++h)
    launch_gemm(stream, U_bf + (size_t)h * B * 1024, 1024,
                Wvw_t + (size_t)h * 256 * 1024, 1024, bvw + h * 256,
                CTX_bf + h * 256, 1024, B, 256, 1024, 0, 1);
  launch_gemm(stream, CTX_bf, 1024, Wow_t, 1024, bow, ATT_bf, 1024, B, 1024, 1024, 0, 1);

  // ================= predictor MLP -> prediction error =====================
  concat_fb_kernel<<<nb(B * 2048), EW, 0, stream>>>(wst, 1024, ATT_bf, 1024, CMB_bf, B * 2048);
  launch_gemm(stream, CMB_bf, 2048, Wp1_t, 2048, bp1, P1H_bf, 1024, B, 1024, 2048, 1, 1);
  launch_gemm(stream, P1H_bf, 1024, Wp2_t, 1024, bp2, PRED, 512, B, 512, 1024, 0, 0);
  sub_bf_kernel<<<nb(B * 512), EW, 0, stream>>>(obs, PRED, ERR_bf, B * 512);
  launch_gemm(stream, ERR_bf, 512, Wep_t, 512, bep, ENC, 1024, B, 1024, 512, 1, 0);

  // ================= internal-memory attention =============================
  launch_gemm(stream, ist_bf, 1024, Wqi_t, 1024, bqi, Q_bf, 1024, B, 1024, 1024, 0, 1);
  for (int h = 0; h < 4; ++h)
    launch_gemm(stream, Q_bf + h * 256, 1024, Wki_c + h * 256, 1024, nullptr,
                U_bf + (size_t)h * B * 1024, 1024, B, 1024, 256, 0, 1);
  attn_mix_kernel<<<B, 128, 0, stream>>>(imem, Q_bf, U_bf, bki, U_bf);
  for (int h = 0; h < 4; ++h)
    launch_gemm(stream, U_bf + (size_t)h * B * 1024, 1024,
                Wvi_t + (size_t)h * 256 * 1024, 1024, bvi + h * 256,
                CTX_bf + h * 256, 1024, B, 256, 1024, 0, 1);
  launch_gemm(stream, CTX_bf, 1024, Woi_t, 1024, boi, ATT_bf, 1024, B, 1024, 1024, 0, 1);

  // ================= internal GRU (update_strength 0.3) ====================
  axpy_bf_kernel<<<nb(B * 1024), EW, 0, stream>>>(ENC, ATT_bf, 0.3f, II_bf, B * 1024);
  launch_gemm(stream, II_bf, 1024, Wii_t, 1024, bii, GI, 3072, B, 3072, 1024, 0, 0);
  launch_gemm(stream, ist_bf, 1024, Whi_t, 1024, bhi, GH, 3072, B, 3072, 1024, 0, 0);
  gru_kernel<<<nb(B * 1024), EW, 0, stream>>>(GI, GH, ist, NWI_bf, B * 1024);

  // ================= time embedding + action head ==========================
  time_emb_kernel<<<nb(B * 128), EW, 0, stream>>>(phase, Wph, bph, Ecyc, cyc, TMB_bf, B * 128);
  concat_bb_kernel<<<nb(B * 1152), EW, 0, stream>>>(NWI_bf, 1024, TMB_bf, 128, AHN_bf, B * 1152);
  launch_gemm(stream, AHN_bf, 1152, Wa1_t, 1152, ba1, HB_bf, 1024, B, 1024, 1152, 1, 1);
  launch_gemm(stream, HB_bf, 1024, Wa2_t, 1024, ba2, (float*)d_out, 64, B, 64, 1024, 1, 0);
}
